// BasicGCN_5411658793081
// MI455X (gfx1250) — matrix operand order repaired and checked by reading, not compiled, not run on hardware
//
#include <hip/hip_runtime.h>
#include <hip/hip_bf16.h>
#include <stdint.h>

typedef __attribute__((ext_vector_type(16))) _Float16 v16h;
typedef __attribute__((ext_vector_type(8)))  float    v8f;

// ---------------------------------------------------------------------------
// Weight pre-pass: W [128x128] f32 row-major (k,n) -> WT [128x128] f16 (n,k)
// ---------------------------------------------------------------------------
__global__ __launch_bounds__(256)
void k_wt_transpose(const float* __restrict__ W, _Float16* __restrict__ WT) {
  int idx = blockIdx.x * 256 + threadIdx.x;   // 16384 elements
  int k = idx >> 7, n = idx & 127;
  WT[n * 128 + k] = (_Float16)W[idx];
}

__global__ __launch_bounds__(256)
void k_fill(float* __restrict__ p, float v, int n) {
  int i = blockIdx.x * 256 + threadIdx.x;
  if (i < n) p[i] = v;
}

__global__ __launch_bounds__(256)
void k_deg(const int* __restrict__ dst, float* __restrict__ deg, int E) {
  int e = blockIdx.x * 256 + threadIdx.x;
  if (e < E) atomicAdd(&deg[dst[e]], 1.0f);
}

__global__ __launch_bounds__(256)
void k_rsqrt(float* __restrict__ d, int n) {
  int i = blockIdx.x * 256 + threadIdx.x;
  if (i < n) d[i] = rsqrtf(d[i]);   // deg >= 1 always (self-loop)
}

// ---------------------------------------------------------------------------
// WMMA GEMM: C[M x 128] = A[M x 128] (f32) @ W[128 x 128], W given as
// WT f16 column-major (WT[n*128 + k]). f32 accumulate.
// Block = 256 thr = 8 waves; wave w owns rows [blk*128 + w*16, +16), all 128
// output columns as 8 16x16 tiles; K-loop of 4 x 32.
// ---------------------------------------------------------------------------
__global__ __launch_bounds__(256)
void k_gemm_wmma(const float* __restrict__ A, const _Float16* __restrict__ WT,
                 float* __restrict__ C, int M) {
  const int wave = threadIdx.x >> 5;
  const int lane = threadIdx.x & 31;
  const int r    = lane & 15;
  const int hi   = lane >> 4;
  const int row0 = blockIdx.x * 128 + wave * 16;
  const int row  = row0 + r;

  v8f acc[8];
#pragma unroll
  for (int c = 0; c < 8; ++c)
    acc[c] = (v8f){0.f, 0.f, 0.f, 0.f, 0.f, 0.f, 0.f, 0.f};

#pragma unroll
  for (int kc = 0; kc < 4; ++kc) {
    // A fragment (16x32 f16): lanes<16 -> K {ka..ka+15} with
    // ka = kc*32; lanes>=16 -> K {ka+16..ka+31} (contiguous, matches B).
    const int ka = kc * 32 + hi * 16;
    float af[16];
    if (row < M) {
      const float4* ap = reinterpret_cast<const float4*>(A + (size_t)row * 128 + ka);
      float4 f0 = ap[0], f1 = ap[1], f2 = ap[2], f3 = ap[3];
      af[0] = f0.x;  af[1] = f0.y;  af[2] = f0.z;  af[3] = f0.w;
      af[4] = f1.x;  af[5] = f1.y;  af[6] = f1.z;  af[7] = f1.w;
      af[8] = f2.x;  af[9] = f2.y;  af[10] = f2.z; af[11] = f2.w;
      af[12] = f3.x; af[13] = f3.y; af[14] = f3.z; af[15] = f3.w;
    } else {
#pragma unroll
      for (int i = 0; i < 16; ++i) af[i] = 0.f;
    }
    v16h a;
#pragma unroll
    for (int i = 0; i < 16; ++i) a[i] = (_Float16)af[i];

    // B fragment (32x16 f16): lane = column n, contiguous 16 K per lane half.
    const int kb = kc * 32 + hi * 16;
#pragma unroll
    for (int c = 0; c < 8; ++c) {
      const int n = c * 16 + r;
      v16h b = *reinterpret_cast<const v16h*>(WT + (size_t)n * 128 + kb);
      acc[c] = __builtin_amdgcn_wmma_f32_16x16x32_f16(
          /*neg_a=*/false, a, /*neg_b=*/false, b,
          /*c_mod=*/(short)0, acc[c], /*reuse_a=*/false, /*reuse_b=*/false);
    }
  }

  // C/D layout: VGPR v, lanes<16 -> M=v, lanes>=16 -> M=v+8; N = lane&15.
#pragma unroll
  for (int c = 0; c < 8; ++c) {
#pragma unroll
    for (int v = 0; v < 8; ++v) {
      int m = row0 + hi * 8 + v;
      if (m < M) C[(size_t)m * 128 + c * 16 + r] = acc[c][v];
    }
  }
}

// ---------------------------------------------------------------------------
// out[i,:] = h[i,:] * dis[i]^2 + bias   (self-loop term + bias init)
// ---------------------------------------------------------------------------
__global__ __launch_bounds__(256)
void k_selfbias(const float* __restrict__ h, const float* __restrict__ dis,
                const float* __restrict__ b, float* __restrict__ out, int n) {
  int i = blockIdx.x * 2 + (threadIdx.x >> 7);
  int t = threadIdx.x & 127;
  if (i < n) {
    float di = dis[i];
    size_t o = (size_t)i * 128 + t;
    out[o] = h[o] * (di * di) + b[t];
  }
}

// ---------------------------------------------------------------------------
// Edge scatter: one wave per edge, float4 per lane (128 features / 32 lanes).
// out[dst,:] += h[src,:] * dis[src]*dis[dst]   via global_atomic_add_f32.
// ---------------------------------------------------------------------------
__global__ __launch_bounds__(256)
void k_scatter(const float* __restrict__ h, const int* __restrict__ src,
               const int* __restrict__ dst, const float* __restrict__ dis,
               float* __restrict__ out, int E) {
  int e = blockIdx.x * 8 + (threadIdx.x >> 5);
  if (e >= E) return;
  int lane = threadIdx.x & 31;
  int s = src[e], d = dst[e];
  float nrm = dis[s] * dis[d];
  float4 hv = reinterpret_cast<const float4*>(h + (size_t)s * 128)[lane];
  float* o = out + (size_t)d * 128 + lane * 4;
  atomicAdd(o + 0, hv.x * nrm);
  atomicAdd(o + 1, hv.y * nrm);
  atomicAdd(o + 2, hv.z * nrm);
  atomicAdd(o + 3, hv.w * nrm);
}

__global__ __launch_bounds__(256)
void k_relu(float* __restrict__ p, int n) {
  int i = blockIdx.x * 256 + threadIdx.x;
  if (i < n) p[i] = fmaxf(p[i], 0.f);
}

__global__ __launch_bounds__(256)
void k_pool(const float* __restrict__ h, const int* __restrict__ batch,
            float* __restrict__ sums, float* __restrict__ cnt, int n) {
  int i = blockIdx.x * 2 + (threadIdx.x >> 7);
  int t = threadIdx.x & 127;
  if (i < n) {
    int g = batch[i];
    atomicAdd(&sums[(size_t)g * 128 + t], h[(size_t)i * 128 + t]);
    if (t == 0) atomicAdd(&cnt[g], 1.0f);
  }
}

__global__ __launch_bounds__(256)
void k_pooldiv(const float* __restrict__ sums, const float* __restrict__ cnt,
               float* __restrict__ pooled, int n) {
  int i = blockIdx.x * 256 + threadIdx.x;
  if (i < n) pooled[i] = sums[i] / fmaxf(cnt[i >> 7], 1.f);
}

__global__ __launch_bounds__(256)
void k_biasrelu(const float* __restrict__ lin, const float* __restrict__ b,
                float* __restrict__ out, int n) {
  int i = blockIdx.x * 256 + threadIdx.x;
  if (i < n) out[i] = fmaxf(lin[i] + b[i & 127], 0.f);
}

// Final head: out[g] = dot(emb[g,:], Wl2[:,0]) + bl2. One wave per graph.
__global__ __launch_bounds__(256)
void k_final(const float* __restrict__ emb, const float* __restrict__ Wl2,
             const float* __restrict__ bl2, float* __restrict__ out, int G) {
  int g = blockIdx.x * 8 + (threadIdx.x >> 5);
  if (g >= G) return;
  int lane = threadIdx.x & 31;
  float4 e4 = reinterpret_cast<const float4*>(emb + (size_t)g * 128)[lane];
  float4 w4 = reinterpret_cast<const float4*>(Wl2)[lane];
  float s = e4.x * w4.x + e4.y * w4.y + e4.z * w4.z + e4.w * w4.w;
#pragma unroll
  for (int off = 16; off > 0; off >>= 1) s += __shfl_xor(s, off, 32);
  if (lane == 0) out[g] = s + bl2[0];
}

// ---------------------------------------------------------------------------
extern "C" void kernel_launch(void* const* d_in, const int* in_sizes, int n_in,
                              void* d_out, int out_size, void* d_ws, size_t ws_size,
                              hipStream_t stream) {
  const float* x   = (const float*)d_in[0];
  const int*   ei  = (const int*)d_in[1];
  const int*   bat = (const int*)d_in[2];
  const float* W1  = (const float*)d_in[3];
  const float* b1  = (const float*)d_in[4];
  const float* W2  = (const float*)d_in[5];
  const float* b2  = (const float*)d_in[6];
  const float* W3  = (const float*)d_in[7];
  const float* b3  = (const float*)d_in[8];
  const float* Wl1 = (const float*)d_in[9];
  const float* bl1 = (const float*)d_in[10];
  const float* Wl2 = (const float*)d_in[11];
  const float* bl2 = (const float*)d_in[12];
  (void)n_in; (void)ws_size;

  const int N = in_sizes[0] / 128;
  const int E = in_sizes[1] / 2;
  const int G = out_size / 129;            // OUT=1, H=128 -> G*(1+128)
  const int* src = ei;
  const int* dst = ei + E;

  auto au = [](size_t v) { return (v + 255) & ~(size_t)255; };
  char* w = (char*)d_ws;
  const size_t featB = au((size_t)N * 128 * sizeof(float));
  float*    P      = (float*)w; w += featB;                       // features
  float*    L      = (float*)w; w += featB;                       // linear out
  float*    dis    = (float*)w; w += au((size_t)N * sizeof(float));
  _Float16* WT     = (_Float16*)w; w += au((size_t)4 * 16384 * sizeof(_Float16));
  float*    sums   = (float*)w; w += au((size_t)G * 128 * sizeof(float));
  float*    cnt    = (float*)w; w += au((size_t)G * sizeof(float));
  float*    pooled = (float*)w; w += au((size_t)G * 128 * sizeof(float));
  float*    embLin = (float*)w; w += au((size_t)G * 128 * sizeof(float));
  float* outG = (float*)d_out;
  float* emb  = outG + G;                  // emb follows out in d_out

  // Weights -> f16 column-major
  k_wt_transpose<<<64, 256, 0, stream>>>(W1,  WT);
  k_wt_transpose<<<64, 256, 0, stream>>>(W2,  WT + 16384);
  k_wt_transpose<<<64, 256, 0, stream>>>(W3,  WT + 32768);
  k_wt_transpose<<<64, 256, 0, stream>>>(Wl1, WT + 49152);

  // Symmetric GCN norm: dis = rsqrt(1 + indegree)
  k_fill<<<(N + 255) / 256, 256, 0, stream>>>(dis, 1.0f, N);
  k_deg<<<(E + 255) / 256, 256, 0, stream>>>(dst, dis, E);
  k_rsqrt<<<(N + 255) / 256, 256, 0, stream>>>(dis, N);

  const int gemmGrid = (N + 127) / 128;
  const float* cur = x;
  const float* bias[3] = {b1, b2, b3};
  for (int l = 0; l < 3; ++l) {
    k_gemm_wmma<<<gemmGrid, 256, 0, stream>>>(cur, WT + l * 16384, L, N);
    k_selfbias<<<(N + 1) / 2, 256, 0, stream>>>(L, dis, bias[l], P, N);
    k_scatter<<<(E + 7) / 8, 256, 0, stream>>>(L, src, dst, dis, P, E);
    k_relu<<<(N * 128 + 255) / 256, 256, 0, stream>>>(P, N * 128);
    cur = P;
  }

  // Global mean pool
  k_fill<<<(G * 128 + 255) / 256, 256, 0, stream>>>(sums, 0.f, G * 128);
  k_fill<<<(G + 255) / 256, 256, 0, stream>>>(cnt, 0.f, G);
  k_pool<<<(N + 1) / 2, 256, 0, stream>>>(P, bat, sums, cnt, N);
  k_pooldiv<<<(G * 128 + 255) / 256, 256, 0, stream>>>(sums, cnt, pooled, G * 128);

  // MLP head
  k_gemm_wmma<<<(G + 127) / 128, 256, 0, stream>>>(pooled, WT + 49152, embLin, G);
  k_biasrelu<<<(G * 128 + 255) / 256, 256, 0, stream>>>(embLin, bl1, emb, G * 128);
  k_final<<<(G + 7) / 8, 256, 0, stream>>>(emb, Wl2, bl2, outG, G);
}
